// GroupAttention_56444460204548
// MI455X (gfx1250) — matrix operand order repaired and checked by reading, not compiled, not run on hardware
//
#include <hip/hip_runtime.h>
#include <hip/hip_bf16.h>
#include <math.h>

// GQA causal flash-attention forward for MI455X (gfx1250, wave32, WMMA).
// B=2, S=2048, 32 q-heads, 8 kv-heads, d=128. fp32 in/out, bf16 tensor ops.
// Double-buffered LDS staging + batched fragment preloads to overlap
// ds/vmem latency with the v_wmma pipe.

typedef __bf16 v16bf __attribute__((ext_vector_type(16)));
typedef float  v8f   __attribute__((ext_vector_type(8)));

#define BATCH        2
#define SEQ          2048
#define NUM_HEADS    32
#define HEAD_SIZE    128
#define NUM_KV_HEADS 8
#define QTILE        128   // query rows per block (8 waves x 16 rows)
#define KTILE        32    // key columns per iteration (K=32 of bf16 WMMA)
#define WAVES        8
#define NEG_FILL     (-50000.0f)
#define KV_STRIDE    (NUM_KV_HEADS * HEAD_SIZE)   // 1024 floats per (b,s) row

__global__ __launch_bounds__(256, 1)
void gqa_flash_fwd(const float* __restrict__ qg,
                   const float* __restrict__ kg,
                   const float* __restrict__ vg,
                   float* __restrict__ outg)
{
    const int tid  = threadIdx.x;
    const int lane = tid & 31;          // wave32
    const int wave = tid >> 5;
    const int half = lane >> 4;         // 0: lanes 0-15, 1: lanes 16-31
    const int nn   = lane & 15;

    // block -> (b, h, q-tile)
    const int qt  = blockIdx.x & 15;            // SEQ/QTILE = 16
    const int h   = (blockIdx.x >> 4) & 31;
    const int b   = blockIdx.x >> 9;
    const int kvh = h >> 2;                     // GROUP = 4

    const int q0 = qt * QTILE;
    const int qw = q0 + wave * 16;              // this wave's first query row

    __shared__ __bf16 k_lds[2][KTILE * HEAD_SIZE];   // [buf][key][dim]
    __shared__ __bf16 v_lds[2][HEAD_SIZE * KTILE];   // [buf][dim][key] (transposed)
    __shared__ __bf16 p_lds[WAVES * 16 * KTILE];     // per-wave P C->A bounce

    const size_t kvbase = ((size_t)(b * SEQ)) * KV_STRIDE + (size_t)kvh * HEAD_SIZE;

    // ---- cooperative K/V tile staging: float4 global loads, fp32 -> bf16
    auto stage = [&](int buf, int j) {
        __bf16* kb = k_lds[buf];
        __bf16* vb = v_lds[buf];
        #pragma unroll
        for (int e = tid * 4; e < KTILE * HEAD_SIZE; e += 256 * 4) {
            const int key = e >> 7;        // / HEAD_SIZE
            const int dim = e & 127;       // multiple of 4
            const size_t src = kvbase + (size_t)(j + key) * KV_STRIDE + dim;
            const float4 k4 = *(const float4*)&kg[src];
            const float4 v4 = *(const float4*)&vg[src];
            kb[key * HEAD_SIZE + dim + 0] = (__bf16)k4.x;
            kb[key * HEAD_SIZE + dim + 1] = (__bf16)k4.y;
            kb[key * HEAD_SIZE + dim + 2] = (__bf16)k4.z;
            kb[key * HEAD_SIZE + dim + 3] = (__bf16)k4.w;
            vb[(dim + 0) * KTILE + key] = (__bf16)v4.x;
            vb[(dim + 1) * KTILE + key] = (__bf16)v4.y;
            vb[(dim + 2) * KTILE + key] = (__bf16)v4.z;
            vb[(dim + 3) * KTILE + key] = (__bf16)v4.w;
        }
    };

    // ---- Q fragments (A-matrix 16x32 bf16 layout, ISA 7.12.2), 4 chunks of K=32
    v16bf qf[4];
    {
        const size_t qbase = ((size_t)(b * SEQ + qw + nn)) * (NUM_HEADS * HEAD_SIZE)
                           + (size_t)h * HEAD_SIZE;
        #pragma unroll
        for (int c = 0; c < 4; ++c) {
            #pragma unroll
            for (int vv = 0; vv < 8; ++vv) {
                const int kk = 32 * c + ((vv & 4) ? 16 : 0) + half * 8 + (vv & 3) * 2;
                qf[c][2 * vv + 0] = (__bf16)qg[qbase + kk + 0];
                qf[c][2 * vv + 1] = (__bf16)qg[qbase + kk + 1];
            }
        }
    }

    // ---- accumulators: O = 16x128 fp32 (8 C-tiles), online-softmax stats
    v8f o[8] = {};
    float m_prev[8], l[8];
    #pragma unroll
    for (int r = 0; r < 8; ++r) { m_prev[r] = -3.0e38f; l[r] = 0.0f; }

    const float scale = 0.08838834764831845f;   // 1/sqrt(128)
    const int ntiles = q0 / KTILE + (QTILE / KTILE);   // causal bound

    stage(0, 0);
    __syncthreads();

    for (int jt = 0; jt < ntiles; ++jt) {
        const int j = jt * KTILE;
        const int cur = jt & 1;

        // stage next tile into the other buffer (overlaps with compute below)
        if (jt + 1 < ntiles) stage(cur ^ 1, j + KTILE);
        if (jt + 2 < ntiles) {   // global_prefetch_b8 two tiles ahead
            const size_t pf = kvbase + (size_t)(j + 2 * KTILE + (tid >> 3)) * KV_STRIDE
                            + (size_t)(tid & 7) * 16;
            __builtin_prefetch(&kg[pf], 0, 1);
            __builtin_prefetch(&vg[pf], 0, 1);
        }

        const __bf16* kbuf = k_lds[cur];
        const __bf16* vbuf = v_lds[cur];

        // ---- preload all 8 QK^T B-fragments (32x16 bf16, lanes 0-15 K=0..7,16..23)
        v16bf kfr[8];
        #pragma unroll
        for (int c = 0; c < 4; ++c) {
            #pragma unroll
            for (int vv = 0; vv < 8; ++vv) {
                const int kk = 32 * c + ((vv & 4) ? 16 : 0) + half * 8 + (vv & 3) * 2;
                kfr[2 * c + 0][2 * vv + 0] = kbuf[nn * HEAD_SIZE + kk + 0];
                kfr[2 * c + 0][2 * vv + 1] = kbuf[nn * HEAD_SIZE + kk + 1];
                kfr[2 * c + 1][2 * vv + 0] = kbuf[(16 + nn) * HEAD_SIZE + kk + 0];
                kfr[2 * c + 1][2 * vv + 1] = kbuf[(16 + nn) * HEAD_SIZE + kk + 1];
            }
        }

        // ---- S = Q * K^T : two 16x16 score tiles, 4 WMMAs each (K=128)
        v8f s0 = {}, s1 = {};
        #pragma unroll
        for (int c = 0; c < 4; ++c) {
            s0 = __builtin_amdgcn_wmma_f32_16x16x32_bf16(false, qf[c], false, kfr[2 * c + 0],
                                                         (short)0, s0, false, false);
            s1 = __builtin_amdgcn_wmma_f32_16x16x32_bf16(false, qf[c], false, kfr[2 * c + 1],
                                                         (short)0, s1, false, false);
        }

        // ---- preload all 8 V B-fragments (overlaps with softmax VALU below)
        v16bf vfr[8];
        #pragma unroll
        for (int t = 0; t < 8; ++t) {
            #pragma unroll
            for (int vv = 0; vv < 8; ++vv) {
                const int kk = ((vv & 4) ? 16 : 0) + half * 8 + (vv & 3) * 2;           // key index within tile
                vfr[t][2 * vv + 0] = vbuf[(16 * t + nn) * KTILE + kk + 0];
                vfr[t][2 * vv + 1] = vbuf[(16 * t + nn) * KTILE + kk + 1];
            }
        }

        // ---- online softmax in C-layout: element r -> row qw + r + 8*half, col nn
        __bf16* pw = &p_lds[wave * 16 * KTILE];
        #pragma unroll
        for (int r = 0; r < 8; ++r) {
            const int qrow = qw + r + 8 * half;
            float a0 = s0[r] * scale;
            float a1 = s1[r] * scale;
            if (j + nn      > qrow) a0 = NEG_FILL;   // causal fill, as in reference
            if (j + 16 + nn > qrow) a1 = NEG_FILL;

            float mx = fmaxf(a0, a1);
            #pragma unroll
            for (int off = 8; off >= 1; off >>= 1)
                mx = fmaxf(mx, __shfl_xor(mx, off, 16));
            const float mnew = fmaxf(m_prev[r], mx);
            const float corr = __expf(m_prev[r] - mnew);
            const float p0 = __expf(a0 - mnew);
            const float p1 = __expf(a1 - mnew);
            float rs = p0 + p1;
            #pragma unroll
            for (int off = 8; off >= 1; off >>= 1)
                rs += __shfl_xor(rs, off, 16);
            l[r] = l[r] * corr + rs;
            m_prev[r] = mnew;
            #pragma unroll
            for (int t = 0; t < 8; ++t) o[t][r] *= corr;

            // P: C-layout -> LDS (bf16), re-read below in A-layout
            pw[(r + 8 * half) * KTILE + nn]      = (__bf16)p0;
            pw[(r + 8 * half) * KTILE + 16 + nn] = (__bf16)p1;
        }
        // LDS ops from the same wave are in-order (ISA ch.5 7.1): store->load safe.

        // ---- O += P * V : P as A-matrix 16x32
        v16bf pa;
        #pragma unroll
        for (int vv = 0; vv < 8; ++vv) {
            const int kk = ((vv & 4) ? 16 : 0) + half * 8 + (vv & 3) * 2;
            pa[2 * vv + 0] = pw[nn * KTILE + kk + 0];
            pa[2 * vv + 1] = pw[nn * KTILE + kk + 1];
        }
        #pragma unroll
        for (int t = 0; t < 8; ++t)
            o[t] = __builtin_amdgcn_wmma_f32_16x16x32_bf16(false, pa, false, vfr[t],
                                                           (short)0, o[t], false, false);

        __syncthreads();   // next-tile staging complete; old buffer reads done
    }

    // ---- epilogue: normalize and store fp32
    #pragma unroll
    for (int r = 0; r < 8; ++r) {
        const float inv = 1.0f / l[r];
        const size_t row = ((size_t)(b * SEQ + qw + r + 8 * half)) * (NUM_HEADS * HEAD_SIZE)
                         + (size_t)h * HEAD_SIZE;
        #pragma unroll
        for (int t = 0; t < 8; ++t)
            outg[row + 16 * t + nn] = o[t][r] * inv;
    }
}

extern "C" void kernel_launch(void* const* d_in, const int* in_sizes, int n_in,
                              void* d_out, int out_size, void* d_ws, size_t ws_size,
                              hipStream_t stream) {
    (void)in_sizes; (void)n_in; (void)out_size; (void)d_ws; (void)ws_size;
    const float* q = (const float*)d_in[0];
    const float* k = (const float*)d_in[1];
    const float* v = (const float*)d_in[2];
    float* out = (float*)d_out;
    const int nblocks = BATCH * NUM_HEADS * (SEQ / QTILE);   // 1024
    gqa_flash_fwd<<<dim3(nblocks), dim3(256), 0, stream>>>(q, k, v, out);
}
